// Encoder_12970801234526
// MI455X (gfx1250) — matrix operand order repaired and checked by reading, not compiled, not run on hardware
//
#include <hip/hip_runtime.h>
#include <hip/hip_bf16.h>

typedef __attribute__((ext_vector_type(16))) _Float16 v16h;
typedef __attribute__((ext_vector_type(8)))  float    v8f;

// ---------------------------------------------------------------------------
// WMMA fragment helpers (CDNA5 16-bit layouts, wave32)
// A (16x32 f16): lanes 0-15 -> M=lane, K = {0..7, 16..23}
//                lanes 16-31 -> M=lane-16, K = {8..15, 24..31}
// B (32x16 f16): lanes 0-15 -> N=lane, K = {0..7, 16..23} ; lanes 16-31 -> N=lane-16, K = {8..15, 24..31}
// C/D (16x16 f32): elem i -> M = (lane>>4)*8 + i, N = lane&15
// ---------------------------------------------------------------------------
__device__ __forceinline__ v16h load_frag_a(const _Float16* A, int ld, int m0, int k0, int lane) {
  int m  = m0 + (lane & 15);
  int kb = k0 + ((lane >> 4) << 3);
  const _Float16* p = A + m * ld + kb;
  v16h a;
#pragma unroll
  for (int j = 0; j < 8; ++j) a[j] = p[j];
#pragma unroll
  for (int j = 0; j < 8; ++j) a[8 + j] = p[16 + j];
  return a;
}

__device__ __forceinline__ v16h load_frag_b(const _Float16* W, int ld, int n0, int k0, int lane) {
  int n  = n0 + (lane & 15);
  int kb = k0 + ((lane >> 4) << 3);
  const _Float16* p = W + n * ld + kb;
  v16h b;
#pragma unroll
  for (int j = 0; j < 8; ++j) b[j] = p[j];
#pragma unroll
  for (int j = 0; j < 8; ++j) b[8 + j] = p[16 + j];
  return b;
}

// One MLP layer: D[Mpad x Cout] = relu(A[Mpad x Kpad] * W^T + bias), f16 out.
// Tile loop index is wave-uniform -> EXEC all-1s around the WMMA.
__device__ __forceinline__ void mlp_layer_wmma(const _Float16* A, _Float16* D, int ldA, int ldD,
                                               const _Float16* __restrict__ W,
                                               const float* __restrict__ bias,
                                               int Mpad, int Kpad, int Cout,
                                               int lane, int wave, int nwaves)
{
  int mt = Mpad >> 4, nt = Cout >> 4;
  for (int t = wave; t < mt * nt; t += nwaves) {
    int m0 = (t / nt) << 4;
    int n0 = (t % nt) << 4;
    v8f acc = {};
    for (int k0 = 0; k0 < Kpad; k0 += 32) {
      v16h a = load_frag_a(A, ldA, m0, k0, lane);
      v16h b = load_frag_b(W, Kpad, n0, k0, lane);
      acc = __builtin_amdgcn_wmma_f32_16x16x32_f16(false, a, false, b,
                                                   (short)0, acc, false, false);
    }
    int n  = n0 + (lane & 15);
    int mb = m0 + ((lane >> 4) << 3);
    float bv = bias[n];
#pragma unroll
    for (int i = 0; i < 8; ++i) {
      float v = acc[i] + bv;
      D[(mb + i) * ldD + n] = (_Float16)(v > 0.f ? v : 0.f);
    }
  }
}

// ---------------------------------------------------------------------------
// Utility kernels
// ---------------------------------------------------------------------------
__global__ void split_pts_kernel(const float* __restrict__ pts,
                                 float* __restrict__ xyz, float* __restrict__ feat, int total) {
  int t = blockIdx.x * blockDim.x + threadIdx.x;
  if (t >= total) return;
  xyz[t * 3 + 0] = pts[t * 4 + 0];
  xyz[t * 3 + 1] = pts[t * 4 + 1];
  xyz[t * 3 + 2] = pts[t * 4 + 2];
  feat[t]        = pts[t * 4 + 3];
}

__global__ void cvt_w_kernel(const float* __restrict__ w, _Float16* __restrict__ w16,
                             int cout, int cin, int kpad) {
  int t = blockIdx.x * blockDim.x + threadIdx.x;
  if (t >= cout * kpad) return;
  int o = t / kpad, k = t - o * kpad;
  w16[t] = (k < cin) ? (_Float16)w[o * cin + k] : (_Float16)0.f;
}

__global__ void copy_f32_kernel(const float* __restrict__ src, float* __restrict__ dst, int n) {
  int t = blockIdx.x * blockDim.x + threadIdx.x;
  if (t < n) dst[t] = src[t];
}

// flat gather: dst[b,s,c] = src[b, idx[b,s], c]
__global__ void gather_kernel(const float* __restrict__ src, const int* __restrict__ idx,
                              float* __restrict__ dst, int N, int S, int C, int total) {
  int t = blockIdx.x * blockDim.x + threadIdx.x;
  if (t >= total) return;
  int c  = t % C;
  int bs = t / C;           // b*S + s
  int b  = bs / S;
  int id = idx[bs];
  dst[t] = src[((size_t)b * N + id) * C + c];
}

// ---------------------------------------------------------------------------
// Farthest point sampling: one block per batch, iterative argmax.
// ---------------------------------------------------------------------------
__global__ void fps_kernel(const float* __restrict__ xyz, int N, int npoint,
                           int* __restrict__ fidx, float* __restrict__ dists) {
  __shared__ float smax[32];
  __shared__ int   sarg[32];
  __shared__ int   s_last;
  int b = blockIdx.x;
  int tid = threadIdx.x, bd = blockDim.x;
  int nwaves = bd >> 5;
  const float* base = xyz + (size_t)b * N * 3;
  float* dd = dists + (size_t)b * N;

  if (tid == 0) { s_last = 0; fidx[b * npoint + 0] = 0; }
  __syncthreads();

  for (int j = 1; j < npoint; ++j) {
    int last = s_last;
    float lx = base[last * 3 + 0];
    float ly = base[last * 3 + 1];
    float lz = base[last * 3 + 2];
    float best = -1.f; int bi = 0;
    for (int i = tid; i < N; i += bd) {
      float dx = base[i * 3 + 0] - lx;
      float dy = base[i * 3 + 1] - ly;
      float dz = base[i * 3 + 2] - lz;
      float d  = dx * dx + dy * dy + dz * dz;
      float od = (j == 1) ? 1e10f : dd[i];
      float nd = od < d ? od : d;
      dd[i] = nd;
      if (nd > best) { best = nd; bi = i; }
    }
    // wave32 reduction
    for (int off = 16; off > 0; off >>= 1) {
      float ob = __shfl_down(best, off, 32);
      int   oi = __shfl_down(bi,   off, 32);
      if (ob > best || (ob == best && oi < bi)) { best = ob; bi = oi; }
    }
    if ((tid & 31) == 0) { smax[tid >> 5] = best; sarg[tid >> 5] = bi; }
    __syncthreads();
    if (tid == 0) {
      float bb = smax[0]; int bj = sarg[0];
      for (int w = 1; w < nwaves; ++w)
        if (smax[w] > bb || (smax[w] == bb && sarg[w] < bj)) { bb = smax[w]; bj = sarg[w]; }
      s_last = bj;
      fidx[b * npoint + j] = bj;
    }
    __syncthreads();
  }
}

// ---------------------------------------------------------------------------
// Ball query: first `ns` in-radius points in index order, pad with first.
// ---------------------------------------------------------------------------
__global__ void ball_query_kernel(const float* __restrict__ xyz, const float* __restrict__ cen,
                                  int N, int S, int ns, float r2, int B,
                                  int* __restrict__ idx) {
  int t = blockIdx.x * blockDim.x + threadIdx.x;
  if (t >= B * S) return;
  int b = t / S;
  const float* base = xyz + (size_t)b * N * 3;
  float cx = cen[t * 3 + 0], cy = cen[t * 3 + 1], cz = cen[t * 3 + 2];
  int* od = idx + (size_t)t * ns;
  int cnt = 0, first = N - 1;
  for (int i = 0; i < N && cnt < ns; ++i) {
    float dx = cx - base[i * 3 + 0];
    float dy = cy - base[i * 3 + 1];
    float dz = cz - base[i * 3 + 2];
    float d = dx * dx + dy * dy + dz * dz;
    if (d < r2) {
      if (cnt == 0) first = i;
      od[cnt++] = i;
    }
  }
  for (int k = cnt; k < ns; ++k) od[k] = first;
}

// ---------------------------------------------------------------------------
// SA layer: group -> up-to-3 WMMA MLP layers -> maxpool. One block per center.
// Ping-pong is statically unrolled so LDS pointers stay addrspace(3)-inferable
// (ds_load_b128 for A fragments) and weights stay global (global_load_b128).
// nlayers is a uniform kernel arg -> uniform branches, EXEC all-1s at WMMA.
// ---------------------------------------------------------------------------
__global__ void sa_group_mlp_kernel(const float* __restrict__ xyz, const float* __restrict__ feat,
                                    const float* __restrict__ cen, const int* __restrict__ idx,
                                    const _Float16* __restrict__ w0, const float* __restrict__ b0,
                                    const _Float16* __restrict__ w1, const float* __restrict__ b1,
                                    const _Float16* __restrict__ w2, const float* __restrict__ b2,
                                    int nlayers, int N, int S, int ns, int Cf,
                                    int k0p, int c0, int k1p, int c1, int k2p, int c2,
                                    int stride, float* __restrict__ out) {
  extern __shared__ _Float16 smem_h[];
  _Float16* buf0 = smem_h;
  _Float16* buf1 = smem_h + ns * stride;

  int tid = threadIdx.x, bd = blockDim.x;
  int lane = tid & 31, wave = tid >> 5, nwaves = bd >> 5;
  int bs = blockIdx.x;          // b*S + s
  int b  = bs / S;

  for (int t = tid; t < 2 * ns * stride; t += bd) smem_h[t] = (_Float16)0.f;
  __syncthreads();

  int cin = 3 + Cf;
  float cx = cen[bs * 3 + 0], cy = cen[bs * 3 + 1], cz = cen[bs * 3 + 2];
  const int* gi = idx + (size_t)bs * ns;
  for (int t = tid; t < ns * cin; t += bd) {
    int k = t / cin, c = t - k * cin;
    int id = gi[k];
    float v;
    if (c == 0)      v = xyz[((size_t)b * N + id) * 3 + 0] - cx;
    else if (c == 1) v = xyz[((size_t)b * N + id) * 3 + 1] - cy;
    else if (c == 2) v = xyz[((size_t)b * N + id) * 3 + 2] - cz;
    else             v = feat[((size_t)b * N + id) * Cf + (c - 3)];
    buf0[k * stride + c] = (_Float16)v;
  }
  __syncthreads();

  // layer 0: buf0 -> buf1
  mlp_layer_wmma(buf0, buf1, stride, stride, w0, b0, ns, k0p, c0, lane, wave, nwaves);
  __syncthreads();
  int clast = c0;

  if (nlayers >= 2) {
    // layer 1: buf1 -> buf0 (re-zero dest so next layer's K-padding cols are 0)
    for (int t = tid; t < ns * stride; t += bd) buf0[t] = (_Float16)0.f;
    __syncthreads();
    mlp_layer_wmma(buf1, buf0, stride, stride, w1, b1, ns, k1p, c1, lane, wave, nwaves);
    __syncthreads();
    clast = c1;
  }
  if (nlayers >= 3) {
    // layer 2: buf0 -> buf1
    for (int t = tid; t < ns * stride; t += bd) buf1[t] = (_Float16)0.f;
    __syncthreads();
    mlp_layer_wmma(buf0, buf1, stride, stride, w2, b2, ns, k2p, c2, lane, wave, nwaves);
    __syncthreads();
    clast = c2;
  }

  // max pool over samples (duplicated per final buffer to keep LDS inference)
  if (nlayers == 2) {
    for (int c = tid; c < clast; c += bd) {
      float mx = -1e30f;
      for (int k = 0; k < ns; ++k) {
        float v = (float)buf0[k * stride + c];
        mx = v > mx ? v : mx;
      }
      out[(size_t)bs * clast + c] = mx;
    }
  } else {
    for (int c = tid; c < clast; c += bd) {
      float mx = -1e30f;
      for (int k = 0; k < ns; ++k) {
        float v = (float)buf1[k * stride + c];
        mx = v > mx ? v : mx;
      }
      out[(size_t)bs * clast + c] = mx;
    }
  }
}

// ---------------------------------------------------------------------------
// Feature propagation: 3-NN interp + 2 WMMA layers + residual add.
// One block per 16 unknown points.
// ---------------------------------------------------------------------------
__global__ void fp_kernel(const float* __restrict__ unknown, const float* __restrict__ known,
                          const float* __restrict__ fskip, const float* __restrict__ fknown,
                          const _Float16* __restrict__ w0, const float* __restrict__ b0,
                          const _Float16* __restrict__ w1, const float* __restrict__ b1,
                          int n, int m, float* __restrict__ out) {
  __shared__ _Float16 bufA[16 * 384];
  __shared__ _Float16 bufB[16 * 384];
  __shared__ float swt[16][3];
  __shared__ int   sid[16][3];
  int tid = threadIdx.x, bd = blockDim.x;
  int lane = tid & 31, wave = tid >> 5, nwaves = bd >> 5;
  int bpb = n / 16;
  int b   = blockIdx.x / bpb;
  int p0  = (blockIdx.x % bpb) * 16;

  if (tid < 16) {
    int p = p0 + tid;
    float ux = unknown[((size_t)b * n + p) * 3 + 0];
    float uy = unknown[((size_t)b * n + p) * 3 + 1];
    float uz = unknown[((size_t)b * n + p) * 3 + 2];
    float d0 = 1e30f, d1 = 1e30f, d2 = 1e30f; int i0 = 0, i1 = 0, i2 = 0;
    for (int j = 0; j < m; ++j) {
      float dx = ux - known[((size_t)b * m + j) * 3 + 0];
      float dy = uy - known[((size_t)b * m + j) * 3 + 1];
      float dz = uz - known[((size_t)b * m + j) * 3 + 2];
      float dd = dx * dx + dy * dy + dz * dz;
      if (dd < d0)      { d2 = d1; i2 = i1; d1 = d0; i1 = i0; d0 = dd; i0 = j; }
      else if (dd < d1) { d2 = d1; i2 = i1; d1 = dd; i1 = j; }
      else if (dd < d2) { d2 = dd; i2 = j; }
    }
    float wa = 1.f / (d0 + 1e-8f), wb = 1.f / (d1 + 1e-8f), wc = 1.f / (d2 + 1e-8f);
    float sw = wa + wb + wc;
    swt[tid][0] = wa / sw; swt[tid][1] = wb / sw; swt[tid][2] = wc / sw;
    sid[tid][0] = i0; sid[tid][1] = i1; sid[tid][2] = i2;
  }
  __syncthreads();

  for (int t = tid; t < 16 * 384; t += bd) {
    int mm = t / 384, c = t - mm * 384;
    float v;
    if (c < 256) {
      v = swt[mm][0] * fknown[((size_t)b * m + sid[mm][0]) * 256 + c]
        + swt[mm][1] * fknown[((size_t)b * m + sid[mm][1]) * 256 + c]
        + swt[mm][2] * fknown[((size_t)b * m + sid[mm][2]) * 256 + c];
    } else {
      v = fskip[((size_t)b * n + p0 + mm) * 128 + (c - 256)];
    }
    bufA[mm * 384 + c] = (_Float16)v;
  }
  __syncthreads();

  mlp_layer_wmma(bufA, bufB, 384, 384, w0, b0, 16, 384, 256, lane, wave, nwaves);
  __syncthreads();
  mlp_layer_wmma(bufB, bufA, 384, 384, w1, b1, 16, 256, 128, lane, wave, nwaves);
  __syncthreads();

  for (int t = tid; t < 16 * 128; t += bd) {
    int mm = t / 128, c = t - mm * 128;
    out[((size_t)b * n + p0 + mm) * 128 + c] =
        (float)bufA[mm * 384 + c] + fskip[((size_t)b * n + p0 + mm) * 128 + c];
  }
}

// ---------------------------------------------------------------------------
// Host orchestration
// ---------------------------------------------------------------------------
extern "C" void kernel_launch(void* const* d_in, const int* in_sizes, int n_in,
                              void* d_out, int out_size, void* d_ws, size_t ws_size,
                              hipStream_t stream) {
  (void)in_sizes; (void)n_in; (void)out_size; (void)ws_size;
  const int B = 2, N0 = 16384;
  const int S1 = 6000, S2 = 2048, S3 = 512, S4 = 256;
  const int NS1 = 48, NS2 = 32, NS3 = 32, NS4 = 16;

  char* wsb = (char*)d_ws;
  size_t off = 0;
  auto alloc = [&](size_t bytes) -> void* {
    void* p = wsb + off;
    off += (bytes + 255) & ~(size_t)255;
    return p;
  };

  float* xyz0  = (float*)alloc((size_t)B * N0 * 3 * 4);
  float* feat0 = (float*)alloc((size_t)B * N0 * 4);
  float* dists = (float*)alloc((size_t)B * N0 * 4);
  int* fidx1 = (int*)alloc((size_t)B * S1 * 4);
  int* fidx2 = (int*)alloc((size_t)B * S2 * 4);
  int* fidx3 = (int*)alloc((size_t)B * S3 * 4);
  int* fidx4 = (int*)alloc((size_t)B * S4 * 4);
  float* xyz1 = (float*)alloc((size_t)B * S1 * 3 * 4);
  float* xyz2 = (float*)alloc((size_t)B * S2 * 3 * 4);
  float* xyz3 = (float*)alloc((size_t)B * S3 * 3 * 4);
  float* xyz4 = (float*)alloc((size_t)B * S4 * 3 * 4);
  float* f1 = (float*)alloc((size_t)B * S1 * 64 * 4);
  float* f2 = (float*)alloc((size_t)B * S2 * 64 * 4);
  float* f3 = (float*)alloc((size_t)B * S3 * 128 * 4);
  float* f4 = (float*)alloc((size_t)B * S4 * 256 * 4);
  int* idx1 = (int*)alloc((size_t)B * S1 * NS1 * 4);
  int* idx2 = (int*)alloc((size_t)B * S2 * NS2 * 4);
  int* idx3 = (int*)alloc((size_t)B * S3 * NS3 * 4);
  int* idx4 = (int*)alloc((size_t)B * S4 * NS4 * 4);

  // f16 weights, K zero-padded to multiple of 32.
  struct WSpec { int in_idx, cout, cin, kpad; _Float16* p; };
  WSpec w[11] = {
    {1, 16, 4, 32, nullptr},  {2, 32, 16, 32, nullptr},  {3, 64, 32, 32, nullptr},
    {7, 64, 67, 96, nullptr}, {8, 64, 64, 64, nullptr},
    {11,128, 67, 96, nullptr},{12,128,128,128, nullptr},
    {15,128,131,160, nullptr},{16,256,128,128, nullptr},{17,256,256,256, nullptr},
    {21,256,384,384, nullptr}
  };
  WSpec wfp1 = {22, 128, 256, 256, nullptr};

  for (int i = 0; i < 11; ++i)
    w[i].p = (_Float16*)alloc((size_t)w[i].cout * w[i].kpad * sizeof(_Float16));
  wfp1.p = (_Float16*)alloc((size_t)wfp1.cout * wfp1.kpad * sizeof(_Float16));

  for (int i = 0; i < 11; ++i) {
    int tot = w[i].cout * w[i].kpad;
    cvt_w_kernel<<<(tot + 255) / 256, 256, 0, stream>>>(
        (const float*)d_in[w[i].in_idx], w[i].p, w[i].cout, w[i].cin, w[i].kpad);
  }
  {
    int tot = wfp1.cout * wfp1.kpad;
    cvt_w_kernel<<<(tot + 255) / 256, 256, 0, stream>>>(
        (const float*)d_in[wfp1.in_idx], wfp1.p, wfp1.cout, wfp1.cin, wfp1.kpad);
  }

  const float* bia[25];
  for (int i = 0; i < 25 && i < n_in; ++i) bia[i] = (const float*)d_in[i];

  // split pts -> xyz0, feat0
  split_pts_kernel<<<(B * N0 + 255) / 256, 256, 0, stream>>>(
      (const float*)d_in[0], xyz0, feat0, B * N0);

  // ---- Level 1 ----
  fps_kernel<<<B, 512, 0, stream>>>(xyz0, N0, S1, fidx1, dists);
  gather_kernel<<<(B * S1 * 3 + 255) / 256, 256, 0, stream>>>(xyz0, fidx1, xyz1, N0, S1, 3, B * S1 * 3);
  ball_query_kernel<<<(B * S1 + 255) / 256, 256, 0, stream>>>(xyz0, xyz1, N0, S1, NS1, 1.6f * 1.6f, B, idx1);
  {
    int stride = 64; size_t sh = (size_t)2 * NS1 * stride * sizeof(_Float16);
    sa_group_mlp_kernel<<<B * S1, 128, sh, stream>>>(
        xyz0, feat0, xyz1, idx1,
        w[0].p, bia[4], w[1].p, bia[5], w[2].p, bia[6],
        3, N0, S1, NS1, 1, 32, 16, 32, 32, 32, 64, stride, f1);
  }

  // ---- Level 2 ----
  fps_kernel<<<B, 512, 0, stream>>>(xyz1, S1, S2, fidx2, dists);
  gather_kernel<<<(B * S2 * 3 + 255) / 256, 256, 0, stream>>>(xyz1, fidx2, xyz2, S1, S2, 3, B * S2 * 3);
  ball_query_kernel<<<(B * S2 + 255) / 256, 256, 0, stream>>>(xyz1, xyz2, S1, S2, NS2, 3.2f * 3.2f, B, idx2);
  {
    int stride = 96; size_t sh = (size_t)2 * NS2 * stride * sizeof(_Float16);
    sa_group_mlp_kernel<<<B * S2, 128, sh, stream>>>(
        xyz1, f1, xyz2, idx2,
        w[3].p, bia[9], w[4].p, bia[10], nullptr, nullptr,
        2, S1, S2, NS2, 64, 96, 64, 64, 64, 0, 0, stride, f2);
  }

  // ---- Level 3 ----
  fps_kernel<<<B, 512, 0, stream>>>(xyz2, S2, S3, fidx3, dists);
  gather_kernel<<<(B * S3 * 3 + 255) / 256, 256, 0, stream>>>(xyz2, fidx3, xyz3, S2, S3, 3, B * S3 * 3);
  ball_query_kernel<<<(B * S3 + 255) / 256, 256, 0, stream>>>(xyz2, xyz3, S2, S3, NS3, 6.4f * 6.4f, B, idx3);
  {
    int stride = 128; size_t sh = (size_t)2 * NS3 * stride * sizeof(_Float16);
    sa_group_mlp_kernel<<<B * S3, 128, sh, stream>>>(
        xyz2, f2, xyz3, idx3,
        w[5].p, bia[13], w[6].p, bia[14], nullptr, nullptr,
        2, S2, S3, NS3, 64, 96, 128, 128, 128, 0, 0, stride, f3);
  }

  // ---- Level 4 ----
  fps_kernel<<<B, 512, 0, stream>>>(xyz3, S3, S4, fidx4, dists);
  gather_kernel<<<(B * S4 * 3 + 255) / 256, 256, 0, stream>>>(xyz3, fidx4, xyz4, S3, S4, 3, B * S4 * 3);
  ball_query_kernel<<<(B * S4 + 255) / 256, 256, 0, stream>>>(xyz3, xyz4, S3, S4, NS4, 10.f * 10.f, B, idx4);
  {
    int stride = 256; size_t sh = (size_t)2 * NS4 * stride * sizeof(_Float16);
    sa_group_mlp_kernel<<<B * S4, 128, sh, stream>>>(
        xyz3, f3, xyz4, idx4,
        w[7].p, bia[18], w[8].p, bia[19], w[9].p, bia[20],
        3, S3, S4, NS4, 128, 160, 128, 128, 256, 256, 256, stride, f4);
  }

  // ---- Feature propagation + residual ----
  float* out_xyz3 = (float*)d_out;                 // 2*512*3
  float* out_ff3  = (float*)d_out + (size_t)B * S3 * 3;
  fp_kernel<<<B * (S3 / 16), 128, 0, stream>>>(
      xyz3, xyz4, f3, f4, w[10].p, bia[23], wfp1.p, bia[24], S3, S4, out_ff3);
  copy_f32_kernel<<<(B * S3 * 3 + 255) / 256, 256, 0, stream>>>(xyz3, out_xyz3, B * S3 * 3);
}
